// MultiHeadTransformerEncoder_1271310320224
// MI455X (gfx1250) — matrix operand order repaired and checked by reading, not compiled, not run on hardware
//
#include <hip/hip_runtime.h>
#include <stdint.h>
#include <stddef.h>

#define D_MODELN 1024
#define NUM_HEADN 16
#define D_KN 64
#define FFN_DN 4096
#define SEQN 2048
#define BATCHN 4
#define TOKENSN (BATCHN * SEQN) /* 8192 */
#define EPSF 1e-5f

typedef __attribute__((ext_vector_type(8)))  __bf16          bf16x8;
typedef __attribute__((ext_vector_type(16))) __bf16          bf16x16;
typedef __attribute__((ext_vector_type(8)))  float           f32x8;
typedef __attribute__((ext_vector_type(4)))  float           f32x4;
typedef __attribute__((ext_vector_type(8)))  unsigned short  u16x8;
typedef __attribute__((ext_vector_type(4)))  unsigned short  u16x4;

__device__ __forceinline__ unsigned short f2bf(float f) {
  union { float f; uint32_t u; } v; v.f = f;
  uint32_t r = v.u + 0x7FFFu + ((v.u >> 16) & 1u);   // round-to-nearest-even
  return (unsigned short)(r >> 16);
}

__device__ __forceinline__ bf16x16 load_frag_pair(const unsigned short* p0,
                                                  const unsigned short* p1) {
  bf16x8 lo = *(const bf16x8*)p0;   // global_load_b128 / ds_load_b128
  bf16x8 hi = *(const bf16x8*)p1;
  return __builtin_shufflevector(lo, hi, 0,1,2,3,4,5,6,7,8,9,10,11,12,13,14,15);
}

__device__ __forceinline__ f32x8 wmma_bf16(bf16x16 a, bf16x16 b, f32x8 c) {
  return __builtin_amdgcn_wmma_f32_16x16x32_bf16(false, a, false, b,
                                                 (short)0, c, false, false);
}

// ---------------------------------------------------------------------------
// fp32 -> bf16 bulk convert (4 elems/thread)
// ---------------------------------------------------------------------------
__global__ __launch_bounds__(256)
void cvt_f32_bf16(const float* __restrict__ in, unsigned short* __restrict__ out, int n4) {
  int i = blockIdx.x * blockDim.x + threadIdx.x;
  if (i < n4) {
    f32x4 v = ((const f32x4*)in)[i];
    u16x4 t;
    t[0] = f2bf(v[0]); t[1] = f2bf(v[1]); t[2] = f2bf(v[2]); t[3] = f2bf(v[3]);
    ((u16x4*)out)[i] = t;
  }
}

// ---------------------------------------------------------------------------
// fp32 [K,N] -> bf16 [N,K] (pre-transposed weights: B-fragments become
// contiguous 128-bit loads: lane = N, two runs of 8 consecutive K values per lane)
// ---------------------------------------------------------------------------
__global__ __launch_bounds__(256)
void cvt_transpose_bf16(const float* __restrict__ w, unsigned short* __restrict__ wt,
                        int K, int N) {
  int i = blockIdx.x * blockDim.x + threadIdx.x;
  if (i < K * N) {
    int k = i / N, n = i - k * N;
    wt[(size_t)n * K + k] = f2bf(w[i]);
  }
}

// ---------------------------------------------------------------------------
// WMMA GEMM: C[M,N] = A[M,K](bf16 row-major) * Wt[N,K](bf16)^T + bias
// One wave -> 32x64 tile: 2 A-frags + 8 accumulators share every B-frag load.
// Weights are L2-resident (<=32 MB each << 192 MB L2), so no LDS staging.
// MODE 0: out bf16 row-major, *oscale   (Q: oscale=1/sqrt(dk); K: 1.0)
// MODE 1: out bf16 transposed [B][N][S] (V projection -> Vt, packed b128 store)
// MODE 2: out fp32 row-major, + resid   (Wo proj, FFN2)
// MODE 3: out bf16 row-major, relu      (FFN1)
// ---------------------------------------------------------------------------
template <int MODE>
__global__ __launch_bounds__(256)
void gemm_wmma(const unsigned short* __restrict__ A,
               const unsigned short* __restrict__ Wt,
               const float* __restrict__ bias,
               const float* __restrict__ resid,
               void* __restrict__ out,
               int M, int N, int K, float oscale) {
  int wave  = blockIdx.x * (blockDim.x >> 5) + (threadIdx.x >> 5);
  int ntile = N >> 6;
  int mt = wave / ntile, nt = wave - mt * ntile;
  if (mt >= (M >> 5)) return;
  int m0 = mt << 5, n0 = nt << 6;
  int lane = threadIdx.x & 31, l = lane & 15, g = lane >> 4;

  f32x8 acc[2][4] = {};
  const unsigned short* arow0 = A + (size_t)(m0 + l) * K;
  const unsigned short* arow1 = A + (size_t)(m0 + 16 + l) * K;

  for (int k = 0; k < K; k += 32) {
    // A fragment: lanes 0-15 -> K chunks {k..k+7, k+16..k+23}; lanes 16-31 +8
    bf16x16 af0 = load_frag_pair(arow0 + k + 8 * g, arow0 + k + 16 + 8 * g);
    bf16x16 af1 = load_frag_pair(arow1 + k + 8 * g, arow1 + k + 16 + 8 * g);
#pragma unroll
    for (int f = 0; f < 4; ++f) {
      const unsigned short* bb = Wt + (size_t)(n0 + f * 16 + l) * K + k;
      bf16x16 bf = load_frag_pair(bb + 8 * g, bb + 16 + 8 * g);      // K chunks {8g..8g+7, 16+8g..16+8g+7}
      acc[0][f] = wmma_bf16(af0, bf, acc[0][f]);
      acc[1][f] = wmma_bf16(af1, bf, acc[1][f]);
    }
  }

#pragma unroll
  for (int hh = 0; hh < 2; ++hh) {
    int mb = m0 + hh * 16;
#pragma unroll
    for (int f = 0; f < 4; ++f) {
      int n = n0 + f * 16 + l;
      float bn = bias[n];
      if (MODE == 0) {
        unsigned short* O = (unsigned short*)out;
#pragma unroll
        for (int i = 0; i < 8; ++i)
          O[(size_t)(mb + i + 8 * g) * N + n] = f2bf((acc[hh][f][i] + bn) * oscale);
      } else if (MODE == 1) {
        // transposed: O[b][n][s]; C-frag rows contiguous in M -> one b128
        unsigned short* O = (unsigned short*)out;
        int b = mb / SEQN, s0 = mb - b * SEQN;
        u16x8 t;
#pragma unroll
        for (int i = 0; i < 8; ++i) t[i] = f2bf(acc[hh][f][i] + bn);
        *(u16x8*)&O[((size_t)b * D_MODELN + n) * SEQN + s0 + 8 * g] = t;
      } else if (MODE == 2) {
        float* O = (float*)out;
#pragma unroll
        for (int i = 0; i < 8; ++i) {
          size_t idx = (size_t)(mb + i + 8 * g) * N + n;
          O[idx] = acc[hh][f][i] + bn + resid[idx];
        }
      } else { // MODE == 3
        unsigned short* O = (unsigned short*)out;
#pragma unroll
        for (int i = 0; i < 8; ++i)
          O[(size_t)(mb + i + 8 * g) * N + n] = f2bf(fmaxf(acc[hh][f][i] + bn, 0.f));
      }
    }
  }
}

// ---------------------------------------------------------------------------
// Flash attention: block = 4 waves, each wave owns 16 query rows of one (b,h).
// Q (pre-scaled by 1/sqrt(dk)) is a natural A-fragment, K row-major is a
// natural B-fragment for Q.K^T, Vt [d_model, seq] is a natural B-fragment
// for P.V.  Raw scores do ONE wave-private LDS round-trip (C->A layout);
// in A layout lane l owns row l, so the softmax max is 15 in-lane v_max +
// a single shfl_xor(16), exp uses a scalar row max, and the bf16 P fragment
// is built directly in registers. Only alpha needs 8 shuffles back into
// C layout for the ctx/l rescale. Row sums come from one WMMA (P @ ones).
// ---------------------------------------------------------------------------
__global__ __launch_bounds__(128)
void attention_wmma(const unsigned short* __restrict__ Qb,
                    const unsigned short* __restrict__ Kb,
                    const unsigned short* __restrict__ Vt,   // [B][1024][2048]
                    unsigned short* __restrict__ Ctx) {      // [tokens][1024]
  __shared__ __align__(16) unsigned short Plds[4][16 * 32];
  int wid  = threadIdx.x >> 5;
  int lane = threadIdx.x & 31, l = lane & 15, g = lane >> 4;

  int bh = blockIdx.x >> 5;            // SEQ/64 = 32 q-chunks per (b,h)
  int qc = blockIdx.x & 31;
  int b  = bh >> 4, h = bh & 15;
  int q0   = qc * 64 + wid * 16;
  int tok0 = b * SEQN + q0;
  const unsigned short* Vb = Vt + (size_t)b * D_MODELN * SEQN;
  unsigned short* P = &Plds[wid][0];

  bf16x16 ones;
#pragma unroll
  for (int i = 0; i < 16; ++i) ones[i] = (__bf16)1.0f;

  // Q A-fragments for d_k = 64 (two k-steps of 32), loaded once
  const unsigned short* qrow = Qb + (size_t)(tok0 + l) * D_MODELN + h * D_KN;
  bf16x16 qa0 = load_frag_pair(qrow + 8 * g,      qrow + 16 + 8 * g);
  bf16x16 qa1 = load_frag_pair(qrow + 32 + 8 * g, qrow + 48 + 8 * g);

  f32x8 ctx[4] = {};
  float lrow[8];
#pragma unroll
  for (int i = 0; i < 8; ++i) lrow[i] = 0.f;
  float mprev = -3.0e38f;              // A-layout running max: this lane's row l

  for (int kc = 0; kc < SEQN; kc += 32) {
    f32x8 s0 = {}, s1 = {};
    {
      const unsigned short* kr0 =
          Kb + (size_t)(b * SEQN + kc + l) * D_MODELN + h * D_KN + 8 * g;
      const unsigned short* kr1 =
          Kb + (size_t)(b * SEQN + kc + 16 + l) * D_MODELN + h * D_KN + 8 * g;
      bf16x16 kb;
      kb = load_frag_pair(kr0,      kr0 + 16); s0 = wmma_bf16(qa0, kb, s0);
      kb = load_frag_pair(kr0 + 32, kr0 + 48); s0 = wmma_bf16(qa1, kb, s0);
      kb = load_frag_pair(kr1,      kr1 + 16); s1 = wmma_bf16(qa0, kb, s1);
      kb = load_frag_pair(kr1 + 32, kr1 + 48); s1 = wmma_bf16(qa1, kb, s1);
    }

    // raw scores (C layout) -> LDS row-major [row][col]
#pragma unroll
    for (int i = 0; i < 8; ++i) {
      P[(i + 8 * g) * 32 + l]      = f2bf(s0[i]);
      P[(i + 8 * g) * 32 + l + 16] = f2bf(s1[i]);
    }
    __builtin_amdgcn_wave_barrier();
    asm volatile("s_wait_dscnt 0" ::: "memory");   // stores visible in-wave

    const unsigned short* pr = P + l * 32;
    bf16x16 sraw = load_frag_pair(pr + 8 * g, pr + 16 + 8 * g);  // A layout
    __builtin_amdgcn_wave_barrier();

    // row l max: 15 in-lane max + one cross-half combine
    float sv[16];
#pragma unroll
    for (int j = 0; j < 16; ++j) sv[j] = (float)sraw[j];
    float mx = sv[0];
#pragma unroll
    for (int j = 1; j < 16; ++j) mx = fmaxf(mx, sv[j]);
    mx = fmaxf(mx, __shfl_xor(mx, 16, 32));
    float mnew   = fmaxf(mprev, mx);
    float alphaA = __expf(mprev - mnew);
    mprev = mnew;

    // P fragment built directly in registers (no second LDS pass)
    bf16x16 pa;
#pragma unroll
    for (int j = 0; j < 16; ++j) pa[j] = (__bf16)__expf(sv[j] - mnew);

    // alpha: A layout (lane = row) -> C layout rows (i + 8g)
    float alphaC[8];
#pragma unroll
    for (int i = 0; i < 8; ++i) alphaC[i] = __shfl(alphaA, 8 * g + i, 16);

    // row sums of P via WMMA: (P @ ones)[m][*] = rowsum(m), already in C layout
    f32x8 srow = {};
    srow = wmma_bf16(pa, ones, srow);
#pragma unroll
    for (int f = 0; f < 4; ++f)
#pragma unroll
      for (int i = 0; i < 8; ++i) ctx[f][i] *= alphaC[i];
#pragma unroll
    for (int i = 0; i < 8; ++i) lrow[i] = lrow[i] * alphaC[i] + srow[i];

#pragma unroll
    for (int f = 0; f < 4; ++f) {
      const unsigned short* vr =
          Vb + (size_t)(h * D_KN + f * 16 + l) * SEQN + kc + 8 * g;
      bf16x16 vb = load_frag_pair(vr, vr + 16);
      ctx[f] = wmma_bf16(pa, vb, ctx[f]);
    }
  }

#pragma unroll
  for (int f = 0; f < 4; ++f) {
    int n = h * D_KN + f * 16 + l;
#pragma unroll
    for (int i = 0; i < 8; ++i)
      Ctx[(size_t)(tok0 + i + 8 * g) * D_MODELN + n] = f2bf(ctx[f][i] / lrow[i]);
  }
}

// ---------------------------------------------------------------------------
// LayerNorm: one wave per 1024-elem row, shuffle reductions, dual output
// (fp32 for next residual, bf16 for next GEMM A-operand).
// ---------------------------------------------------------------------------
__global__ __launch_bounds__(256)
void layernorm_row(const float* __restrict__ X, const float* __restrict__ gamma,
                   const float* __restrict__ beta, float* __restrict__ Of,
                   unsigned short* __restrict__ Ob, int rows) {
  int wave = blockIdx.x * (blockDim.x >> 5) + (threadIdx.x >> 5);
  if (wave >= rows) return;
  int lane = threadIdx.x & 31;
  const float* x = X + (size_t)wave * D_MODELN;

  f32x4 v[8];
  float s = 0.f;
#pragma unroll
  for (int j = 0; j < 8; ++j) {
    v[j] = *(const f32x4*)(x + j * 128 + lane * 4);
    s += v[j][0] + v[j][1] + v[j][2] + v[j][3];
  }
  s += __shfl_xor(s, 1, 32);  s += __shfl_xor(s, 2, 32);
  s += __shfl_xor(s, 4, 32);  s += __shfl_xor(s, 8, 32);
  s += __shfl_xor(s, 16, 32);
  float mean = s * (1.f / D_MODELN);

  float ss = 0.f;
#pragma unroll
  for (int j = 0; j < 8; ++j)
#pragma unroll
    for (int t = 0; t < 4; ++t) { float d = v[j][t] - mean; ss += d * d; }
  ss += __shfl_xor(ss, 1, 32);  ss += __shfl_xor(ss, 2, 32);
  ss += __shfl_xor(ss, 4, 32);  ss += __shfl_xor(ss, 8, 32);
  ss += __shfl_xor(ss, 16, 32);
  float rstd = rsqrtf(ss * (1.f / D_MODELN) + EPSF);

#pragma unroll
  for (int j = 0; j < 8; ++j) {
    int base = j * 128 + lane * 4;
    f32x4 gm = *(const f32x4*)(gamma + base);
    f32x4 bt = *(const f32x4*)(beta + base);
    f32x4 y;
#pragma unroll
    for (int t = 0; t < 4; ++t) y[t] = (v[j][t] - mean) * rstd * gm[t] + bt[t];
    *(f32x4*)(Of + (size_t)wave * D_MODELN + base) = y;
    if (Ob) {
      u16x4 yb;
#pragma unroll
      for (int t = 0; t < 4; ++t) yb[t] = f2bf(y[t]);
      *(u16x4*)(Ob + (size_t)wave * D_MODELN + base) = yb;
    }
  }
}

// ---------------------------------------------------------------------------
extern "C" void kernel_launch(void* const* d_in, const int* in_sizes, int n_in,
                              void* d_out, int out_size, void* d_ws, size_t ws_size,
                              hipStream_t stream) {
  const float* x   = (const float*)d_in[0];
  const float* wq  = (const float*)d_in[1];
  const float* bq  = (const float*)d_in[2];
  const float* wk  = (const float*)d_in[3];
  const float* bk  = (const float*)d_in[4];
  const float* wv  = (const float*)d_in[5];
  const float* bv  = (const float*)d_in[6];
  const float* wo  = (const float*)d_in[7];
  const float* bo  = (const float*)d_in[8];
  const float* w1  = (const float*)d_in[9];
  const float* b1  = (const float*)d_in[10];
  const float* w2  = (const float*)d_in[11];
  const float* b2  = (const float*)d_in[12];
  const float* g1  = (const float*)d_in[13];
  const float* be1 = (const float*)d_in[14];
  const float* g2  = (const float*)d_in[15];
  const float* be2 = (const float*)d_in[16];
  float* out = (float*)d_out;

  uint8_t* ws = (uint8_t*)d_ws;
  size_t off = 0;
  auto alloc = [&](size_t bytes) -> void* {
    void* p = ws + off;
    off += (bytes + 255) & ~(size_t)255;
    return p;
  };
  const size_t TD  = (size_t)TOKENSN * D_MODELN;
  unsigned short* xb  = (unsigned short*)alloc(TD * 2);
  unsigned short* wqt = (unsigned short*)alloc((size_t)D_MODELN * D_MODELN * 2);
  unsigned short* wkt = (unsigned short*)alloc((size_t)D_MODELN * D_MODELN * 2);
  unsigned short* wvt = (unsigned short*)alloc((size_t)D_MODELN * D_MODELN * 2);
  unsigned short* wot = (unsigned short*)alloc((size_t)D_MODELN * D_MODELN * 2);
  unsigned short* w1t = (unsigned short*)alloc((size_t)D_MODELN * FFN_DN * 2);
  unsigned short* w2t = (unsigned short*)alloc((size_t)FFN_DN * D_MODELN * 2);
  unsigned short* Qb  = (unsigned short*)alloc(TD * 2);
  unsigned short* KbB = (unsigned short*)alloc(TD * 2);
  unsigned short* VtB = (unsigned short*)alloc(TD * 2);
  unsigned short* Ctx = (unsigned short*)alloc(TD * 2);
  float*          S1  = (float*)alloc(TD * 4);       // reused as S2
  float*          x1f = (float*)alloc(TD * 4);
  unsigned short* x1b = (unsigned short*)alloc(TD * 2);
  unsigned short* Hb  = (unsigned short*)alloc((size_t)TOKENSN * FFN_DN * 2);

  // 1) convert activations + transpose-convert weights to bf16
  cvt_f32_bf16<<<(TD / 4 + 255) / 256, 256, 0, stream>>>(x, xb, (int)(TD / 4));
  int mm = D_MODELN * D_MODELN;
  cvt_transpose_bf16<<<(mm + 255) / 256, 256, 0, stream>>>(wq, wqt, D_MODELN, D_MODELN);
  cvt_transpose_bf16<<<(mm + 255) / 256, 256, 0, stream>>>(wk, wkt, D_MODELN, D_MODELN);
  cvt_transpose_bf16<<<(mm + 255) / 256, 256, 0, stream>>>(wv, wvt, D_MODELN, D_MODELN);
  cvt_transpose_bf16<<<(mm + 255) / 256, 256, 0, stream>>>(wo, wot, D_MODELN, D_MODELN);
  int mf = D_MODELN * FFN_DN;
  cvt_transpose_bf16<<<(mf + 255) / 256, 256, 0, stream>>>(w1, w1t, D_MODELN, FFN_DN);
  cvt_transpose_bf16<<<(mf + 255) / 256, 256, 0, stream>>>(w2, w2t, FFN_DN, D_MODELN);

  // 2) Q/K/V projections; Q pre-scaled by 1/sqrt(dk) (exact power of two),
  //    V stored transposed for the P.V B-fragments
  int qkv_blocks = (TOKENSN / 32) * (D_MODELN / 64) / 8;   // 512
  gemm_wmma<0><<<qkv_blocks, 256, 0, stream>>>(xb, wqt, bq, nullptr, Qb,
                                               TOKENSN, D_MODELN, D_MODELN, 0.125f);
  gemm_wmma<0><<<qkv_blocks, 256, 0, stream>>>(xb, wkt, bk, nullptr, KbB,
                                               TOKENSN, D_MODELN, D_MODELN, 1.0f);
  gemm_wmma<1><<<qkv_blocks, 256, 0, stream>>>(xb, wvt, bv, nullptr, VtB,
                                               TOKENSN, D_MODELN, D_MODELN, 1.0f);

  // 3) flash attention
  attention_wmma<<<BATCHN * NUM_HEADN * (SEQN / 64), 128, 0, stream>>>(Qb, KbB, VtB, Ctx);

  // 4) output projection + residual, then LN1 (fp32 + bf16 outputs)
  gemm_wmma<2><<<qkv_blocks, 256, 0, stream>>>(Ctx, wot, bo, x, S1,
                                               TOKENSN, D_MODELN, D_MODELN, 1.0f);
  layernorm_row<<<TOKENSN / 8, 256, 0, stream>>>(S1, g1, be1, x1f, x1b, TOKENSN);

  // 5) FFN
  int ffn1_blocks = (TOKENSN / 32) * (FFN_DN / 64) / 8;    // 2048
  gemm_wmma<3><<<ffn1_blocks, 256, 0, stream>>>(x1b, w1t, b1, nullptr, Hb,
                                                TOKENSN, FFN_DN, D_MODELN, 1.0f);
  gemm_wmma<2><<<qkv_blocks, 256, 0, stream>>>(Hb, w2t, b2, x1f, S1,
                                               TOKENSN, D_MODELN, FFN_DN, 1.0f);

  // 6) final LayerNorm -> fp32 output
  layernorm_row<<<TOKENSN / 8, 256, 0, stream>>>(S1, g2, be2, out, nullptr, TOKENSN);

  (void)in_sizes; (void)n_in; (void)out_size; (void)ws_size;
}
